// GcnEncoderBranch_26001732010448
// MI455X (gfx1250) — matrix operand order repaired and checked by reading, not compiled, not run on hardware
//
#include <hip/hip_runtime.h>
#include <hip/hip_bf16.h>

// Problem constants (match reference)
#define BATCH 32
#define WINF  128   // node feature dim F (win_size)
#define NNODE 1024  // number of nodes N (d_model)
#define DGDIM 64    // graph embed dim
#define TOPK  15

typedef __attribute__((ext_vector_type(16))) _Float16 v16h;
typedef __attribute__((ext_vector_type(8)))  float    v8f;

// K index inside a 16-element A fragment (contiguous order, matches B fills):
//   lanes 0-15 (hi=0): K=0..15
//   lanes16-31 (hi=1): K=16..31
__device__ __forceinline__ int a_kidx(int e, int hi) {
    return e + (hi << 4);
}

// Async copy: 16 contiguous global bytes -> 16 contiguous LDS bytes (per lane).
// Tracked by ASYNCcnt (CDNA5 ISA).
__device__ __forceinline__ void async_copy_b128(unsigned lds_addr, const void* gaddr) {
    asm volatile("global_load_async_to_lds_b128 %0, %1, off"
                 :: "v"(lds_addr), "v"((unsigned long long)(size_t)gaddr)
                 : "memory");
}
__device__ __forceinline__ void wait_async0() {
    asm volatile("s_wait_asynccnt 0x0" ::: "memory");
}

// ---------------------------------------------------------------------------
// Kernel 1: q = nf@Wq.T + bq ; k = nf@Wk.T + bk      (per batch: 1024x128x64)
// grid = BATCH*64 (16-node tiles), block = 128 (4 waves, one per 16-wide DG slice)
// x tile staged through LDS with async b128 copies (feature-major layout).
// q/k stored as f16 (they feed f16 WMMAs in kernel 2 anyway).
// ---------------------------------------------------------------------------
__global__ __launch_bounds__(128)
void qk_kernel(const float* __restrict__ x,
               const float* __restrict__ Wq, const float* __restrict__ bq,
               const float* __restrict__ Wk, const float* __restrict__ bk,
               _Float16* __restrict__ qout, _Float16* __restrict__ kout)
{
    __shared__ float xt[WINF][16];       // xt[f][node-in-tile], 8 KB
    const int blk = blockIdx.x;
    const int b   = blk >> 6;
    const int n0  = (blk & 63) << 4;
    const int t   = threadIdx.x;

    const float* xb = x + (size_t)b * WINF * NNODE + n0;   // x[b, f, n0+...]
    const unsigned lbase = (unsigned)(size_t)(&xt[0][0]);
    {
        const int i0 = (t & 3) << 2;     // 0,4,8,12
        const int fb = t >> 2;           // 0..31
        #pragma unroll
        for (int rep = 0; rep < 4; ++rep) {
            int f = rep * 32 + fb;
            async_copy_b128(lbase + (unsigned)((f * 16 + i0) * 4),
                            xb + (size_t)f * NNODE + i0);
        }
    }
    wait_async0();
    __syncthreads();

    const int lane = t & 31;
    const int wv   = t >> 5;              // 0..3
    const int col  = lane & 15;
    const int hi   = lane >> 4;
    const int d0   = wv << 4;

    v8f cq = {}; v8f ck = {};
    for (int c = 0; c < 4; ++c) {         // K = 128 in 4 chunks of 32
        const int f0 = c << 5;
        v16h a, bqv, bkv;
        #pragma unroll
        for (int e = 0; e < 16; ++e)
            a[e] = (_Float16)xt[f0 + a_kidx(e, hi)][col];
        const float* wqr = Wq + (d0 + col) * WINF + f0 + (hi << 4);
        const float* wkr = Wk + (d0 + col) * WINF + f0 + (hi << 4);
        #pragma unroll
        for (int e = 0; e < 16; ++e) {
            bqv[e] = (_Float16)wqr[e];
            bkv[e] = (_Float16)wkr[e];
        }
        cq = __builtin_amdgcn_wmma_f32_16x16x32_f16(false, a, false, bqv, (short)0, cq, false, false);
        ck = __builtin_amdgcn_wmma_f32_16x16x32_f16(false, a, false, bkv, (short)0, ck, false, false);
    }

    const float biasq = bq[d0 + col];
    const float biask = bk[d0 + col];
    #pragma unroll
    for (int r = 0; r < 8; ++r) {
        int n = n0 + r + (hi << 3);
        size_t o = ((size_t)b * NNODE + n) * DGDIM + d0 + col;
        qout[o] = (_Float16)(cq[r] + biasq);
        kout[o] = (_Float16)(ck[r] + biask);
    }
}

// ---------------------------------------------------------------------------
// Kernel 2: scores = (q @ k.T)/8, diag=-inf, per-row top-15 + degree counts
// grid = BATCH*64 (16-row tiles), block = 512 (16 waves)
// waves 0..15 each compute 4 column tiles (WMMA) into 16x1024 LDS, then
// wave w extracts top-15 of row w via shfl argmax.
// ---------------------------------------------------------------------------
__global__ __launch_bounds__(512)
void score_topk_kernel(const _Float16* __restrict__ qin, const _Float16* __restrict__ kin,
                       int* __restrict__ topk, unsigned* __restrict__ deg)
{
    __shared__ float sc[16][NNODE];      // 64 KB
    const int blk  = blockIdx.x;
    const int b    = blk >> 6;
    const int n0   = (blk & 63) << 4;
    const int t    = threadIdx.x;
    const int wv   = t >> 5;             // 0..15
    const int lane = t & 31;
    const int col  = lane & 15;
    const int hi   = lane >> 4;

    const _Float16* qb = qin + (size_t)b * NNODE * DGDIM;
    const _Float16* kb = kin + (size_t)b * NNODE * DGDIM;

    // A fragments: rows q[n0..n0+15], K = 64 (two 32-chunks)
    v16h a0, a1;
    #pragma unroll
    for (int e = 0; e < 16; ++e) {
        a0[e] = qb[(n0 + col) * DGDIM +      a_kidx(e, hi)];
        a1[e] = qb[(n0 + col) * DGDIM + 32 + a_kidx(e, hi)];
    }

    for (int mt = 0; mt < 4; ++mt) {
        const int m0 = ((wv << 2) + mt) << 4;
        const _Float16* kr = kb + (m0 + col) * DGDIM + (hi << 4);
        if (mt < 3) __builtin_prefetch(kr + 16 * DGDIM, 0, 0);   // next m-tile
        v16h b0, b1;
        #pragma unroll
        for (int e = 0; e < 16; ++e) {
            b0[e] = kr[e];         // K = e + 16*hi
            b1[e] = kr[32 + e];    // K = 32 + e + 16*hi
        }
        v8f acc = {};
        acc = __builtin_amdgcn_wmma_f32_16x16x32_f16(false, a0, false, b0, (short)0, acc, false, false);
        acc = __builtin_amdgcn_wmma_f32_16x16x32_f16(false, a1, false, b1, (short)0, acc, false, false);
        #pragma unroll
        for (int r = 0; r < 8; ++r) {
            int rl = r + (hi << 3);
            float v = acc[r] * 0.125f;                 // 1/sqrt(64)
            if (n0 + rl == m0 + col) v = -__builtin_inff();
            sc[rl][m0 + col] = v;
        }
    }
    __syncthreads();

    // wave wv: top-15 of row wv (values live in registers, strided by lane)
    const int row = wv;
    float vals[32];
    #pragma unroll
    for (int j = 0; j < 32; ++j) vals[j] = sc[row][lane + (j << 5)];

    int*      tkrow = topk + ((size_t)b * NNODE + n0 + row) * TOPK;
    unsigned* degb  = deg + (size_t)b * NNODE;

    for (int it = 0; it < TOPK; ++it) {
        float best = -__builtin_inff(); int bj = 0;
        #pragma unroll
        for (int j = 0; j < 32; ++j)
            if (vals[j] > best) { best = vals[j]; bj = j; }
        int bidx = (bj << 5) + lane;
        #pragma unroll
        for (int off = 16; off > 0; off >>= 1) {
            float ov = __shfl_xor(best, off, 32);
            int   oi = __shfl_xor(bidx, off, 32);
            if (ov > best || (ov == best && oi < bidx)) { best = ov; bidx = oi; }
        }
        if ((bidx & 31) == lane) vals[bidx >> 5] = -__builtin_inff();
        if (lane == 0) {
            tkrow[it] = bidx;
            atomicAdd(&degb[bidx], 1u);
        }
    }
}

// ---------------------------------------------------------------------------
// Kernel 3: h = nf @ Wg.T      (per batch: 1024x128x128)
// grid = BATCH*64, block = 256 (8 waves, one per 16-wide out-feature slice)
// ---------------------------------------------------------------------------
__global__ __launch_bounds__(256)
void h_kernel(const float* __restrict__ x, const float* __restrict__ Wg,
              float* __restrict__ hout)
{
    __shared__ float xt[WINF][16];       // xt[f][node-in-tile], 8 KB
    const int blk = blockIdx.x;
    const int b   = blk >> 6;
    const int n0  = (blk & 63) << 4;
    const int t   = threadIdx.x;

    const float* xb = x + (size_t)b * WINF * NNODE + n0;
    const unsigned lbase = (unsigned)(size_t)(&xt[0][0]);
    {
        const int i0 = (t & 3) << 2;     // 0,4,8,12
        const int fb = t >> 2;           // 0..63
        #pragma unroll
        for (int rep = 0; rep < 2; ++rep) {
            int f = rep * 64 + fb;
            async_copy_b128(lbase + (unsigned)((f * 16 + i0) * 4),
                            xb + (size_t)f * NNODE + i0);
        }
    }
    wait_async0();
    __syncthreads();

    const int lane = t & 31;
    const int wv   = t >> 5;             // 0..7
    const int col  = lane & 15;
    const int hi   = lane >> 4;
    const int fo0  = wv << 4;

    v8f acc = {};
    for (int c = 0; c < 4; ++c) {
        const int f0 = c << 5;
        v16h a, bv;
        #pragma unroll
        for (int e = 0; e < 16; ++e)
            a[e] = (_Float16)xt[f0 + a_kidx(e, hi)][col];
        const float* wr = Wg + (fo0 + col) * WINF + f0 + (hi << 4);
        #pragma unroll
        for (int e = 0; e < 16; ++e)
            bv[e] = (_Float16)wr[e];
        acc = __builtin_amdgcn_wmma_f32_16x16x32_f16(false, a, false, bv, (short)0, acc, false, false);
    }
    #pragma unroll
    for (int r = 0; r < 8; ++r) {
        int n = n0 + r + (hi << 3);
        hout[((size_t)b * NNODE + n) * WINF + fo0 + col] = acc[r];
    }
}

// ---------------------------------------------------------------------------
// Kernel 4a: degree init (2 self loops per node)
// ---------------------------------------------------------------------------
__global__ void init_deg(unsigned* __restrict__ deg)
{
    int i = blockIdx.x * 256 + threadIdx.x;
    if (i < BATCH * NNODE) deg[i] = 2u;
}

// ---------------------------------------------------------------------------
// Kernel 4b: out[b,f,n] = bg[f] + (2/deg[n]) * h[b,n,f]   (bias + self loops)
// ---------------------------------------------------------------------------
__global__ __launch_bounds__(256)
void init_out(const float* __restrict__ hin, const unsigned* __restrict__ deg,
              const float* __restrict__ bg, float* __restrict__ out)
{
    size_t idx = (size_t)blockIdx.x * 256 + threadIdx.x;   // < B*WINF*NNODE
    int b   = (int)(idx >> 17);          // WINF*NNODE = 131072
    int rem = (int)(idx & 131071);
    int f   = rem >> 10;                 // NNODE = 1024
    int n   = rem & 1023;
    float d  = (float)deg[b * NNODE + n];
    out[idx] = bg[f] + (2.0f / d) * hin[((size_t)b * NNODE + n) * WINF + f];
}

// ---------------------------------------------------------------------------
// Kernel 4c: edge scatter: out[b,f,tgt] += rsqrt(deg[src])*rsqrt(deg[tgt])*h[b,src,f]
// grid = BATCH*NNODE (one block per source node), block = 128 (f)
// ---------------------------------------------------------------------------
__global__ __launch_bounds__(128)
void scatter_kernel(const float* __restrict__ hin, const unsigned* __restrict__ deg,
                    const int* __restrict__ topk, float* __restrict__ out)
{
    const int blk = blockIdx.x;
    const int b   = blk >> 10;
    const int src = blk & 1023;
    const int f   = threadIdx.x;

    const unsigned* degb = deg + (size_t)b * NNODE;
    const float dsrc = rsqrtf((float)degb[src]);
    const float hv   = hin[((size_t)b * NNODE + src) * WINF + f] * dsrc;
    const int*  tk   = topk + ((size_t)b * NNODE + src) * TOPK;
    float* outb = out + (size_t)b * WINF * NNODE + (size_t)f * NNODE;

    #pragma unroll
    for (int kk = 0; kk < TOPK; ++kk) {
        int   tgt = tk[kk];
        float w   = rsqrtf((float)degb[tgt]);
        unsafeAtomicAdd(&outb[tgt], hv * w);
    }
}

// ---------------------------------------------------------------------------
extern "C" void kernel_launch(void* const* d_in, const int* in_sizes, int n_in,
                              void* d_out, int out_size, void* d_ws, size_t ws_size,
                              hipStream_t stream) {
    const float* x  = (const float*)d_in[0];
    const float* Wq = (const float*)d_in[1];
    const float* bq = (const float*)d_in[2];
    const float* Wk = (const float*)d_in[3];
    const float* bk = (const float*)d_in[4];
    const float* Wg = (const float*)d_in[5];
    const float* bg = (const float*)d_in[6];
    float* out = (float*)d_out;

    char* ws = (char*)d_ws;
    const size_t qBytes   = (size_t)BATCH * NNODE * DGDIM * sizeof(_Float16); // 4 MB
    const size_t hBytes   = (size_t)BATCH * NNODE * WINF * sizeof(float);     // 16 MB
    const size_t degBytes = (size_t)BATCH * NNODE * sizeof(unsigned);         // 128 KB
    _Float16* qbuf = (_Float16*)(ws);
    _Float16* kbuf = (_Float16*)(ws + qBytes);
    float*    hbuf = (float*)(ws + 2 * qBytes);
    unsigned* dbuf = (unsigned*)(ws + 2 * qBytes + hBytes);
    int*      tbuf = (int*)(ws + 2 * qBytes + hBytes + degBytes);

    const int tiles = BATCH * (NNODE / 16);                 // 2048

    init_deg<<<(BATCH * NNODE + 255) / 256, 256, 0, stream>>>(dbuf);
    qk_kernel<<<tiles, 128, 0, stream>>>(x, Wq, bq, Wk, bk, qbuf, kbuf);
    score_topk_kernel<<<tiles, 512, 0, stream>>>(qbuf, kbuf, tbuf, dbuf);
    h_kernel<<<tiles, 256, 0, stream>>>(x, Wg, hbuf);
    init_out<<<(BATCH * WINF * NNODE) / 256, 256, 0, stream>>>(hbuf, dbuf, bg, out);
    scatter_kernel<<<BATCH * NNODE, 128, 0, stream>>>(hbuf, dbuf, tbuf, out);
}
